// MultiHead_Attention_22033182228929
// MI455X (gfx1250) — matrix operand order repaired and checked by reading, compiled, not run on hardware
//
#include <hip/hip_runtime.h>
#include <hip/hip_bf16.h>
#include <stdint.h>

// MHA dims (compile-time, match reference)
#define Bv 4
#define Nv 2048
#define Cv 1024
#define Hv 16
#define Dv 64
#define KSPLIT 2
#define SSTRIDE 20   // padded h-stride for score LDS (16B aligned, bank-friendly)

typedef __attribute__((ext_vector_type(16))) __bf16 v16bf;
typedef __attribute__((ext_vector_type(4)))  __bf16 v4bf;
typedef __attribute__((ext_vector_type(8)))  float  v8f;

union Frag16 { v16bf v; uint4 u[2]; };

// ---- WMMA wrapper: D = A(16x32 bf16) * B(32x16 bf16) + C(16x16 f32) ----
static __device__ __forceinline__ v8f wmma_bf16(v16bf a, v16bf b, v8f c) {
  return __builtin_amdgcn_wmma_f32_16x16x32_bf16(
      /*neg_a=*/false, a, /*neg_b=*/false, b,
      /*c_mod=*/(short)0, c, /*reuse_a=*/false, /*reuse_b=*/false);
}

// A-fragment (16x32, bf16 source): lane holds row M=lane%16,
// K = off..off+7 and off+16..off+23 with off=(lane/16)*8.
// p points at (row, K=off); two disjoint 16B loads.
static __device__ __forceinline__ v16bf load_frag_a_bf16(const __bf16* p) {
  Frag16 f;
  f.u[0] = *(const uint4*)(p);
  f.u[1] = *(const uint4*)(p + 16);
  return f.v;
}

// B-fragment (32x16, bf16 source): lane holds col N=lane%16,
// K = off..off+7 and off+16..off+23 with off=(lane/16)*8.
static __device__ __forceinline__ v16bf load_frag_b_bf16(const __bf16* p) {
  Frag16 f;
  f.u[0] = *(const uint4*)(p);
  f.u[1] = *(const uint4*)(p + 16);
  return f.v;
}

// A-fragment built from f32 source (convert to bf16 on the fly).
static __device__ __forceinline__ v16bf load_frag_a_f32(const float* p) {
  float4 f0 = *(const float4*)(p);
  float4 f1 = *(const float4*)(p + 4);
  float4 f2 = *(const float4*)(p + 16);
  float4 f3 = *(const float4*)(p + 20);
  v16bf a;
  a[0]  = (__bf16)f0.x; a[1]  = (__bf16)f0.y; a[2]  = (__bf16)f0.z; a[3]  = (__bf16)f0.w;
  a[4]  = (__bf16)f1.x; a[5]  = (__bf16)f1.y; a[6]  = (__bf16)f1.z; a[7]  = (__bf16)f1.w;
  a[8]  = (__bf16)f2.x; a[9]  = (__bf16)f2.y; a[10] = (__bf16)f2.z; a[11] = (__bf16)f2.w;
  a[12] = (__bf16)f3.x; a[13] = (__bf16)f3.y; a[14] = (__bf16)f3.z; a[15] = (__bf16)f3.w;
  return a;
}

// A-fragment = (p[] + q[]) converted to bf16 (sums the two ctx k-split partials).
static __device__ __forceinline__ v16bf load_frag_a_f32_sum(const float* p, const float* q) {
  float4 a0 = *(const float4*)(p);      float4 b0 = *(const float4*)(q);
  float4 a1 = *(const float4*)(p + 4);  float4 b1 = *(const float4*)(q + 4);
  float4 a2 = *(const float4*)(p + 16); float4 b2 = *(const float4*)(q + 16);
  float4 a3 = *(const float4*)(p + 20); float4 b3 = *(const float4*)(q + 20);
  v16bf a;
  a[0]  = (__bf16)(a0.x + b0.x); a[1]  = (__bf16)(a0.y + b0.y);
  a[2]  = (__bf16)(a0.z + b0.z); a[3]  = (__bf16)(a0.w + b0.w);
  a[4]  = (__bf16)(a1.x + b1.x); a[5]  = (__bf16)(a1.y + b1.y);
  a[6]  = (__bf16)(a1.z + b1.z); a[7]  = (__bf16)(a1.w + b1.w);
  a[8]  = (__bf16)(a2.x + b2.x); a[9]  = (__bf16)(a2.y + b2.y);
  a[10] = (__bf16)(a2.z + b2.z); a[11] = (__bf16)(a2.w + b2.w);
  a[12] = (__bf16)(a3.x + b3.x); a[13] = (__bf16)(a3.y + b3.y);
  a[14] = (__bf16)(a3.z + b3.z); a[15] = (__bf16)(a3.w + b3.w);
  return a;
}

// ---------------- Kernel 0: transpose + f32->bf16 convert the 4 weights ----
// wt[z][n][k] = w_z[k][n]  (so B-fragments read contiguous K runs)
__global__ void __launch_bounds__(256)
wt_transpose_kernel(const float* __restrict__ wq, const float* __restrict__ wk,
                    const float* __restrict__ wv, const float* __restrict__ wo,
                    __bf16* __restrict__ wt) {
  __shared__ float tile[32][33];
  const float* src = (blockIdx.z == 0) ? wq : (blockIdx.z == 1) ? wk
                   : (blockIdx.z == 2) ? wv : wo;
  __bf16* dst = wt + (size_t)blockIdx.z * (size_t)(Cv * Cv);
  int tx = threadIdx.x, ty = threadIdx.y;
  int i0 = blockIdx.y * 32, j0 = blockIdx.x * 32;
#pragma unroll
  for (int r = 0; r < 4; r++)
    tile[ty + r * 8][tx] = src[(size_t)(i0 + ty + r * 8) * Cv + j0 + tx];
  __syncthreads();
#pragma unroll
  for (int r = 0; r < 4; r++)
    dst[(size_t)(j0 + ty + r * 8) * Cv + i0 + tx] = (__bf16)tile[tx][ty + r * 8];
}

// ---------------- Kernel 1: QKV projection (x f32 -> Q,K,V bf16) ----------
// One wave = 16 rows x 64 cols, 3 outputs simultaneously (A reuse).
// Q,K stored [b*n][c] bf16 row-major; V stored transposed vt[b][c][n] bf16.
__global__ void __launch_bounds__(128)
qkv_proj_kernel(const float* __restrict__ x,
                const __bf16* __restrict__ wtq, const __bf16* __restrict__ wtk,
                const __bf16* __restrict__ wtv,
                __bf16* __restrict__ qws, __bf16* __restrict__ kws,
                __bf16* __restrict__ vtws) {
  int tid = threadIdx.x, lane = tid & 31;
  int l16 = lane & 15, lh = lane >> 4;
  int wid = blockIdx.x * 4 + (tid >> 5);
  int mt = wid >> 4, ng = wid & 15;
  int m0 = mt * 16, n0 = ng * 64;

  v8f z = {0.f, 0.f, 0.f, 0.f, 0.f, 0.f, 0.f, 0.f};
  v8f acc[3][4];
#pragma unroll
  for (int w = 0; w < 3; w++)
#pragma unroll
    for (int nt = 0; nt < 4; nt++) acc[w][nt] = z;

  const __bf16* wts[3] = {wtq, wtk, wtv};
  for (int k0 = 0; k0 < Cv; k0 += 32) {
    v16bf a = load_frag_a_f32(x + (size_t)(m0 + l16) * Cv + k0 + lh * 8);
#pragma unroll
    for (int w = 0; w < 3; w++) {
#pragma unroll
      for (int nt = 0; nt < 4; nt++) {
        v16bf bf = load_frag_b_bf16(wts[w] + (size_t)(n0 + nt * 16 + l16) * Cv +
                                    k0 + lh * 8);
        acc[w][nt] = wmma_bf16(a, bf, acc[w][nt]);
      }
    }
  }

  int b = m0 >> 11;        // 2048 rows per batch
  int nseq0 = m0 & 2047;
#pragma unroll
  for (int nt = 0; nt < 4; nt++) {
#pragma unroll
    for (int i = 0; i < 8; i++) {
      int ml = i + lh * 8;           // D-layout: m = r + 8*(lane/16)
      int col = n0 + nt * 16 + l16;  //           n = lane%16
      size_t rowoff = (size_t)(m0 + ml) * Cv + col;
      qws[rowoff] = (__bf16)acc[0][nt][i];
      kws[rowoff] = (__bf16)acc[1][nt][i];
      vtws[(size_t)b * Cv * Nv + (size_t)col * Nv + (nseq0 + ml)] =
          (__bf16)acc[2][nt][i];
    }
  }
}

// ---------------- Kernel 2: fused attention with head-axis softmax --------
// WG = 4 waves = 128 threads, owns (b, q-tile of 16, k-split half).
// Wave w handles heads [4w, 4w+4). Scores for all 16 heads of a (q,k) tile
// meet in LDS; softmax over h=16 is elementwise; ctx accumulated in WMMA.
__global__ void __launch_bounds__(128)
attention_kernel(const __bf16* __restrict__ qws, const __bf16* __restrict__ kws,
                 const __bf16* __restrict__ vtws,
                 float* __restrict__ ctx0, float* __restrict__ ctx1) {
  __shared__ float  sS[512 * SSTRIDE]; // [elem e=q*32+k][h], stride 20 (pad)
  __shared__ __bf16 sA[16 * 512];      // attn bf16, [h][q][k0..31]

  int tid = threadIdx.x, lane = tid & 31, wv = tid >> 5;
  int l16 = lane & 15, lh = lane >> 4;
  int wg = blockIdx.x;
  int ks = wg & (KSPLIT - 1);
  int qt = (wg >> 1) & 127;
  int b  = wg >> 8;
  int q0 = qt * 16;
  int h0 = wv * 4;
  float* cdst = ks ? ctx1 : ctx0;

  // Preload Q A-fragments for this wave's 4 heads (2 d-chunks of 32 each).
  v16bf qf[4][2];
#pragma unroll
  for (int hh = 0; hh < 4; hh++)
#pragma unroll
    for (int dc = 0; dc < 2; dc++)
      qf[hh][dc] = load_frag_a_bf16(qws + (size_t)(b * Nv + q0 + l16) * Cv +
                                    (h0 + hh) * Dv + dc * 32 + lh * 8);

  v8f z = {0.f, 0.f, 0.f, 0.f, 0.f, 0.f, 0.f, 0.f};
  v8f cacc[4][4];
#pragma unroll
  for (int hh = 0; hh < 4; hh++)
#pragma unroll
    for (int dt = 0; dt < 4; dt++) cacc[hh][dt] = z;

  const int CHUNKS = (Nv / KSPLIT) / 32;
  const int kbase0 = ks * (Nv / KSPLIT);

  for (int ch = 0; ch < CHUNKS; ch++) {
    int kb  = kbase0 + ch * 32;
    int kbn = kbase0 + ((ch + 1 < CHUNKS) ? (ch + 1) : ch) * 32;
    // Prefetch next chunk's K rows and V^T rows (global_prefetch_b8).
    __builtin_prefetch(kws + (size_t)(b * Nv + kbn + l16) * Cv + h0 * Dv, 0, 1);
    __builtin_prefetch(vtws + (size_t)b * Cv * Nv +
                           (size_t)(h0 * Dv + l16) * Nv + kbn, 0, 1);

    // ---- scores: s[h][q][k] = Q·K^T over d=64 (2 WMMA per 16x16 tile) ----
    v8f s[4][2];
#pragma unroll
    for (int hh = 0; hh < 4; hh++) { s[hh][0] = z; s[hh][1] = z; }
#pragma unroll
    for (int ksub = 0; ksub < 2; ksub++)
#pragma unroll
      for (int hh = 0; hh < 4; hh++)
#pragma unroll
        for (int dc = 0; dc < 2; dc++) {
          v16bf kf = load_frag_b_bf16(
              kws + (size_t)(b * Nv + kb + ksub * 16 + l16) * Cv +
              (h0 + hh) * Dv + dc * 32 + lh * 8);
          s[hh][ksub] = wmma_bf16(qf[hh][dc], kf, s[hh][ksub]);
        }

    // ---- scatter scaled scores to LDS; 4 owned heads are h-adjacent so
    //      each (elem) writes one float4 (ds_store_b128, bank-clean) -----
#pragma unroll
    for (int ksub = 0; ksub < 2; ksub++)
#pragma unroll
      for (int i = 0; i < 8; i++) {
        int ml = i + lh * 8;
        int e  = ml * 32 + ksub * 16 + l16;
        float4 v4;
        v4.x = s[0][ksub][i] * 0.125f;   // 1/sqrt(64)
        v4.y = s[1][ksub][i] * 0.125f;
        v4.z = s[2][ksub][i] * 0.125f;
        v4.w = s[3][ksub][i] * 0.125f;
        *(float4*)&sS[e * SSTRIDE + h0] = v4;
      }
    __syncthreads();

    // ---- softmax over the 16 heads (reference uses axis=1 = heads) ----
    // Each thread owns 4 consecutive (q,k) elements of one q-row segment.
    int e0  = tid * 4;
    int qq  = e0 >> 5, kk0 = e0 & 31;
    float pv[4][16];
    float inv[4];
#pragma unroll
    for (int r = 0; r < 4; r++) {
      int e = e0 + r;
      float4 p0 = *(const float4*)&sS[e * SSTRIDE + 0];
      float4 p1 = *(const float4*)&sS[e * SSTRIDE + 4];
      float4 p2 = *(const float4*)&sS[e * SSTRIDE + 8];
      float4 p3 = *(const float4*)&sS[e * SSTRIDE + 12];
      pv[r][0]=p0.x;  pv[r][1]=p0.y;  pv[r][2]=p0.z;  pv[r][3]=p0.w;
      pv[r][4]=p1.x;  pv[r][5]=p1.y;  pv[r][6]=p1.z;  pv[r][7]=p1.w;
      pv[r][8]=p2.x;  pv[r][9]=p2.y;  pv[r][10]=p2.z; pv[r][11]=p2.w;
      pv[r][12]=p3.x; pv[r][13]=p3.y; pv[r][14]=p3.z; pv[r][15]=p3.w;
      float mx = -1e30f;
#pragma unroll
      for (int h = 0; h < 16; h++) mx = fmaxf(mx, pv[r][h]);
      float sum = 0.f;
#pragma unroll
      for (int h = 0; h < 16; h++) { pv[r][h] = __expf(pv[r][h] - mx); sum += pv[r][h]; }
      inv[r] = 1.0f / sum;
    }
    // 4 consecutive k positions per head -> one 8B bf16x4 store per head.
#pragma unroll
    for (int h = 0; h < 16; h++) {
      v4bf w;
      w[0] = (__bf16)(pv[0][h] * inv[0]);
      w[1] = (__bf16)(pv[1][h] * inv[1]);
      w[2] = (__bf16)(pv[2][h] * inv[2]);
      w[3] = (__bf16)(pv[3][h] * inv[3]);
      *(v4bf*)&sA[h * 512 + qq * 32 + kk0] = w;
    }
    __syncthreads();

    // ---- ctx[h][q][d] += attn[h] (16x32) x V[h] (32x16 per d-tile) ----
#pragma unroll
    for (int hh = 0; hh < 4; hh++) {
      v16bf af = load_frag_a_bf16(sA + (h0 + hh) * 512 + l16 * 32 + lh * 8);
#pragma unroll
      for (int dt = 0; dt < 4; dt++) {
        v16bf vf = load_frag_b_bf16(
            vtws + (size_t)b * Cv * Nv +
            (size_t)((h0 + hh) * Dv + dt * 16 + l16) * Nv + kb + lh * 8);
        cacc[hh][dt] = wmma_bf16(af, vf, cacc[hh][dt]);
      }
    }
    __syncthreads();  // protect sS/sA before next chunk overwrites
  }

  // ---- write this k-split's partial ctx (layout [b][q][h*64+d], f32) ----
#pragma unroll
  for (int hh = 0; hh < 4; hh++)
#pragma unroll
    for (int dt = 0; dt < 4; dt++)
#pragma unroll
      for (int i = 0; i < 8; i++) {
        int ml = i + lh * 8;
        cdst[((size_t)b * Nv + q0 + ml) * Cv + (h0 + hh) * Dv + dt * 16 + l16] =
            cacc[hh][dt][i];
      }
}

// ---------------- Kernel 3: output projection (sum ctx partials) ----------
__global__ void __launch_bounds__(128)
out_proj_kernel(const float* __restrict__ c0, const float* __restrict__ c1,
                const __bf16* __restrict__ wto, float* __restrict__ out) {
  int tid = threadIdx.x, lane = tid & 31;
  int l16 = lane & 15, lh = lane >> 4;
  int wid = blockIdx.x * 4 + (tid >> 5);
  int mt = wid >> 4, ng = wid & 15;
  int m0 = mt * 16, n0 = ng * 64;

  v8f z = {0.f, 0.f, 0.f, 0.f, 0.f, 0.f, 0.f, 0.f};
  v8f acc[4] = {z, z, z, z};

  for (int k0 = 0; k0 < Cv; k0 += 32) {
    size_t aoff = (size_t)(m0 + l16) * Cv + k0 + lh * 8;
    v16bf a = load_frag_a_f32_sum(c0 + aoff, c1 + aoff);
#pragma unroll
    for (int nt = 0; nt < 4; nt++) {
      v16bf bf = load_frag_b_bf16(wto + (size_t)(n0 + nt * 16 + l16) * Cv +
                                  k0 + lh * 8);
      acc[nt] = wmma_bf16(a, bf, acc[nt]);
    }
  }
#pragma unroll
  for (int nt = 0; nt < 4; nt++)
#pragma unroll
    for (int i = 0; i < 8; i++) {
      int ml = i + lh * 8;
      out[(size_t)(m0 + ml) * Cv + n0 + nt * 16 + l16] = acc[nt][i];
    }
}

// ---------------- host launcher ----------------
extern "C" void kernel_launch(void* const* d_in, const int* in_sizes, int n_in,
                              void* d_out, int out_size, void* d_ws, size_t ws_size,
                              hipStream_t stream) {
  (void)in_sizes; (void)n_in; (void)out_size; (void)ws_size;
  const float* x  = (const float*)d_in[0];
  const float* wq = (const float*)d_in[1];
  const float* wk = (const float*)d_in[2];
  const float* wv = (const float*)d_in[3];
  const float* wo = (const float*)d_in[4];

  char* ws = (char*)d_ws;
  // Workspace map (120 MB total):
  //   [0,   8MB)  wt: 4 transposed bf16 weights (1M elems each)
  //   [8,  24MB)  q_ws  bf16 [b*n][c]
  //   [24, 40MB)  k_ws  bf16 [b*n][c]
  //   [40, 56MB)  vt_ws bf16 [b][c][n]
  //   [56, 88MB)  ctx0  f32  [b*n][c]  (k-split 0 partial)
  //   [88,120MB)  ctx1  f32  [b*n][c]  (k-split 1 partial)
  __bf16* wt   = (__bf16*)ws;
  __bf16* qws  = (__bf16*)(ws + (size_t)(8u  << 20));
  __bf16* kws  = (__bf16*)(ws + (size_t)(24u << 20));
  __bf16* vtws = (__bf16*)(ws + (size_t)(40u << 20));
  float*  ctx0 = (float*) (ws + (size_t)(56u << 20));
  float*  ctx1 = (float*) (ws + (size_t)(88u << 20));

  wt_transpose_kernel<<<dim3(32, 32, 4), dim3(32, 8, 1), 0, stream>>>(
      wq, wk, wv, wo, wt);
  qkv_proj_kernel<<<2048, 128, 0, stream>>>(
      x, wt, wt + (size_t)(1u << 20), wt + (size_t)(2u << 20), qws, kws, vtws);
  attention_kernel<<<1024, 128, 0, stream>>>(qws, kws, vtws, ctx0, ctx1);
  out_proj_kernel<<<2048, 128, 0, stream>>>(
      ctx0, ctx1, wt + (size_t)(3u << 20), (float*)d_out);
}
